// Model_39676907886177
// MI455X (gfx1250) — matrix operand order repaired and checked by reading, not compiled, not run on hardware
//
#include <hip/hip_runtime.h>
#include <hip/hip_bf16.h>

// MI455X / gfx1250, wave32. Fused QKV projection + MHA via BF16 WMMA
// (v_wmma_f32_16x16x32_bf16, fp32 accumulate) + async-to-LDS staging
// (global_load_async_to_lds_b128 / s_wait_asynccnt, double-buffered).
//
// Pipeline:
//   k0: convert X (query) and W to bf16 once (packed cvt), into workspace
//   k1: QKV = Xbf * Wbf^T + bias ; route Q->[BH][S][dh], K->[BH][S][dh],
//       V^T->[BH][dh][S] (bf16) so every stage-2 fragment load is 128-bit
//   k2: flash attention, one wave per (b,h,32-query block): K/V tiles
//       double-buffered into LDS via the async engine, fragments via
//       ds_load_b128, online softmax in fp32.
//
// Workspace (bytes): 3*8MB (Q,K,Vt) + 8MB (Xbf) + 6MB (Wbf) ~= 39.8 MB

#define BATCH  2
#define SEQ    2048
#define DMODEL 1024
#define NH     16
#define DH     64
#define ROWS   (BATCH * SEQ)   // 4096
#define NCOLS  (3 * DMODEL)    // 3072

typedef __attribute__((ext_vector_type(16))) __bf16 v16bf;
typedef __attribute__((ext_vector_type(8)))  float  v8f;
typedef __attribute__((ext_vector_type(2)))  float  vf2;
typedef __attribute__((ext_vector_type(2)))  __bf16 vbf2;
typedef __attribute__((ext_vector_type(4)))  int    v4i;

union Frag {                       // one WMMA A/B operand: 16 bf16 in 8 dwords
    v16bf        v;
    unsigned int u[8];
};

// packed fp32x2 -> bf16x2 (RNE)
__device__ __forceinline__ unsigned int cvt2(float a, float b) {
    vf2 f; f.x = a; f.y = b;
    vbf2 h = __builtin_convertvector(f, vbf2);
    union { vbf2 v; unsigned int u; } c; c.v = h; return c.u;
}
__device__ __forceinline__ unsigned short bf1(float f) {
    union { __bf16 b; unsigned short u; } c; c.b = (__bf16)f; return c.u;
}

// A-matrix 16x32 bf16: lanes 0-15 hold K=0..15, lanes 16-31 K=16..31 (row=lane&15).
// B-matrix 32x16 bf16: lanes 0-15 hold K=0..15, lanes 16-31 K=16..31 (col=lane&15).
// Both are two 16-byte contiguous loads per lane. Templated on pointer type so
// address-space inference keeps LDS accesses as ds_load_b128 after inlining.
template <typename P>
__device__ __forceinline__ void load_frag2(Frag& f, P p) {
    f.u[0] = ((const unsigned int*)p)[0]; f.u[1] = ((const unsigned int*)p)[1];
    f.u[2] = ((const unsigned int*)p)[2]; f.u[3] = ((const unsigned int*)p)[3];
    f.u[4] = ((const unsigned int*)p)[4]; f.u[5] = ((const unsigned int*)p)[5];
    f.u[6] = ((const unsigned int*)p)[6]; f.u[7] = ((const unsigned int*)p)[7];
}
template <typename P>
__device__ __forceinline__ void load_fragA(Frag& f, P row, int kbase, int half) {
    const unsigned int* lo = (const unsigned int*)(row + kbase + half * 16);
    const unsigned int* hi = (const unsigned int*)(row + kbase + half * 16 + 8);
    f.u[0] = lo[0]; f.u[1] = lo[1]; f.u[2] = lo[2]; f.u[3] = lo[3];
    f.u[4] = hi[0]; f.u[5] = hi[1]; f.u[6] = hi[2]; f.u[7] = hi[3];
}

// ---- CDNA5 async copy: global -> LDS, 16B per lane, ASYNCcnt-tracked ------
// Builtin signature (from probe diagnostic): (v4i AS1*, v4i AS3*, imm, imm)
__device__ __forceinline__ void async_b128(const unsigned short* g, unsigned short* l) {
#if __has_builtin(__builtin_amdgcn_global_load_async_to_lds_b128)
    __builtin_amdgcn_global_load_async_to_lds_b128(
        (__attribute__((address_space(1))) v4i*)g,
        (__attribute__((address_space(3))) v4i*)l, 0, 0);
#else
    unsigned int loff =
        (unsigned int)(size_t)(__attribute__((address_space(3))) unsigned short*)l;
    asm volatile("global_load_async_to_lds_b128 %0, %1, off"
                 :: "v"(loff), "v"(g) : "memory");
#endif
}
template <int N>
__device__ __forceinline__ void wait_asynccnt() {
#if __has_builtin(__builtin_amdgcn_s_wait_asynccnt)
    __builtin_amdgcn_s_wait_asynccnt(N);
#else
    asm volatile("s_wait_asynccnt %0" :: "i"(N) : "memory");
#endif
}

// ---------------------------------------------------------------------------
// Kernel 0: fp32 -> bf16 bulk conversion (float4 -> 2 dwords per thread)
// ---------------------------------------------------------------------------
__global__ __launch_bounds__(256) void cvt_bf16_kernel(
    const float* __restrict__ src, unsigned short* __restrict__ dst, int n4)
{
    int i = blockIdx.x * blockDim.x + threadIdx.x;
    if (i < n4) {
        float4 f = ((const float4*)src)[i];
        uint2 o; o.x = cvt2(f.x, f.y); o.y = cvt2(f.z, f.w);
        ((uint2*)dst)[i] = o;
    }
}

// ---------------------------------------------------------------------------
// Kernel 1: QKV = Xbf * Wbf^T + bias.  One wave per 32(M) x 64(N) tile:
// 2 A-frags x 4 B-frags -> 8 WMMAs per 12 b128 loads per K-step.
// ---------------------------------------------------------------------------
__global__ __launch_bounds__(256) void qkv_proj_kernel(
    const unsigned short* __restrict__ Xbf, const unsigned short* __restrict__ Wbf,
    const float* __restrict__ bias,
    unsigned short* __restrict__ wsq, unsigned short* __restrict__ wsk,
    unsigned short* __restrict__ wsvt)
{
    const int lane = threadIdx.x & 31;
    const int wid  = (blockIdx.x * blockDim.x + threadIdx.x) >> 5;
    const int mt   = wid / (NCOLS / 64);            // 0..127  (32-row tiles)
    const int ng   = wid - mt * (NCOLS / 64);       // 0..47   (64-col groups)
    const int row0 = mt * 32;
    const int c0   = ng * 64;
    const int half = (lane >> 4) & 1;
    const int lm   = lane & 15;

    v8f acc0[4], acc1[4];
    #pragma unroll
    for (int t = 0; t < 4; ++t)
        #pragma unroll
        for (int r = 0; r < 8; ++r) { acc0[t][r] = 0.0f; acc1[t][r] = 0.0f; }

    const unsigned short* xrow0 = Xbf + (size_t)(row0 + lm) * DMODEL;
    const unsigned short* xrow1 = Xbf + (size_t)(row0 + 16 + lm) * DMODEL;

    for (int k0 = 0; k0 < DMODEL; k0 += 32) {
        Frag a0, a1;
        load_fragA(a0, xrow0, k0, half);
        load_fragA(a1, xrow1, k0, half);
        #pragma unroll
        for (int t = 0; t < 4; ++t) {
            const unsigned short* wrow =
                Wbf + (size_t)(c0 + t * 16 + lm) * DMODEL + k0 + half * 16;
            Frag b;
            load_frag2(b, wrow);
            acc0[t] = __builtin_amdgcn_wmma_f32_16x16x32_bf16(
                false, a0.v, false, b.v, (short)0, acc0[t], false, false);
            acc1[t] = __builtin_amdgcn_wmma_f32_16x16x32_bf16(
                false, a1.v, false, b.v, (short)0, acc1[t], false, false);
        }
    }

    // Epilogue: bias add, route to Q / K / V^T workspace (bf16).
    #pragma unroll
    for (int t = 0; t < 4; ++t) {
        const int col   = c0 + t * 16 + lm;
        const float bv  = bias[col];
        const int which = col >> 10;             // 0=Q 1=K 2=V (uniform per tile)
        const int dcol  = col & (DMODEL - 1);
        const int h  = dcol >> 6;
        const int dd = dcol & (DH - 1);
        #pragma unroll
        for (int m2 = 0; m2 < 2; ++m2) {
            const v8f* acct = m2 ? &acc1[t] : &acc0[t];
            #pragma unroll
            for (int r = 0; r < 8; ++r) {
                const int row = row0 + m2 * 16 + r + 8 * half;
                const int bb  = row >> 11;
                const int s   = row & (SEQ - 1);
                const int bh  = bb * NH + h;
                const unsigned short e = bf1((*acct)[r] + bv);
                if (which == 0)      wsq [((size_t)bh * SEQ + s) * DH + dd] = e;
                else if (which == 1) wsk [((size_t)bh * SEQ + s) * DH + dd] = e;
                else                 wsvt[((size_t)bh * DH + dd) * SEQ + s] = e;
            }
        }
    }
}

// ---------------------------------------------------------------------------
// Kernel 2: flash attention. One wave per (b, h, 32-query block); two 16-row
// query tiles share every K/V fragment. K/V tiles double-buffered into LDS
// via the async engine. scores = (Q.K^T) * 32 (faithful inverted scaling).
// ---------------------------------------------------------------------------
#define KSTRIDE 72   // 144 B rows: 16B-aligned, 36-bank stride (conflict-free)
#define VSTRIDE 40   // 80 B rows:  16B-aligned, 20-bank stride (conflict-free)
#define PROW    40

__global__ __launch_bounds__(32) void attn_kernel(
    const unsigned short* __restrict__ wsq, const unsigned short* __restrict__ wsk,
    const unsigned short* __restrict__ wsvt, float* __restrict__ out)
{
    __shared__ unsigned short ldsK[2][32 * KSTRIDE]; // 32 keys x 64 dh
    __shared__ unsigned short ldsV[2][64 * VSTRIDE]; // 64 dh  x 32 keys
    __shared__ unsigned short ldsP[16 * PROW];       // 16 x 32 P tile

    const int lane = threadIdx.x & 31;
    const int half = lane >> 4;
    const int lm   = lane & 15;
    const int bh   = blockIdx.x >> 6;            // 0..31
    const int q0   = (blockIdx.x & 63) * 32;
    const int b    = bh >> 4;
    const int h    = bh & 15;

    const unsigned short* kbase = wsk  + (size_t)bh * SEQ * DH;
    const unsigned short* vbase = wsvt + (size_t)bh * DH * SEQ;

    // 16 async b128 ops stage one K tile + one V^T tile into buffer `buf`.
    auto stage = [&](int buf, int j0s) {
        const unsigned short* kg = kbase + (size_t)(j0s + lane) * DH;
        #pragma unroll
        for (int i = 0; i < 8; ++i)
            async_b128(kg + i * 8, &ldsK[buf][lane * KSTRIDE + i * 8]);
        #pragma unroll
        for (int i = 0; i < 8; ++i) {
            const int seg = i * 32 + lane;
            const int row = seg >> 2, c = (seg & 3) * 8;
            async_b128(vbase + (size_t)row * SEQ + j0s + c,
                       &ldsV[buf][row * VSTRIDE + c]);
        }
    };

    // Persistent Q A-fragments: 2 query tiles x (dh 0..31, 32..63)
    Frag aq[2][2];
    #pragma unroll
    for (int qa = 0; qa < 2; ++qa) {
        const unsigned short* qrow =
            wsq + ((size_t)bh * SEQ + q0 + qa * 16 + lm) * DH;
        load_fragA(aq[qa][0], qrow, 0,  half);
        load_fragA(aq[qa][1], qrow, 32, half);
    }

    v8f acc[2][4];
    float mrow[2][8], lrow[2][8];
    #pragma unroll
    for (int qa = 0; qa < 2; ++qa)
        #pragma unroll
        for (int r = 0; r < 8; ++r) {
            mrow[qa][r] = -3.0e38f; lrow[qa][r] = 0.0f;
            acc[qa][0][r] = acc[qa][1][r] = acc[qa][2][r] = acc[qa][3][r] = 0.0f;
        }

    stage(0, 0);                                   // prime buffer 0

    for (int j0 = 0; j0 < SEQ; j0 += 32) {
        const int buf = (j0 >> 5) & 1;
        if (j0 + 32 < SEQ) {                       // prefetch next tile, then
            stage(buf ^ 1, j0 + 32);               // wait only for current one
            wait_asynccnt<16>();                   // (async loads retire in order)
        } else {
            wait_asynccnt<0>();
        }

        // K B-fragments from LDS, shared by both query tiles.
        Frag bk[2][2];                             // [key tile][dh half]
        #pragma unroll
        for (int kt = 0; kt < 2; ++kt) {
            const int rb = (kt * 16 + lm) * KSTRIDE + half * 16;
            load_frag2(bk[kt][0], &ldsK[buf][rb]);
            load_frag2(bk[kt][1], &ldsK[buf][rb + 32]);
        }

        #pragma unroll
        for (int qa = 0; qa < 2; ++qa) {
            v8f s0, s1;
            #pragma unroll
            for (int r = 0; r < 8; ++r) { s0[r] = 0.0f; s1[r] = 0.0f; }
            s0 = __builtin_amdgcn_wmma_f32_16x16x32_bf16(false, aq[qa][0].v, false, bk[0][0].v, (short)0, s0, false, false);
            s0 = __builtin_amdgcn_wmma_f32_16x16x32_bf16(false, aq[qa][1].v, false, bk[0][1].v, (short)0, s0, false, false);
            s1 = __builtin_amdgcn_wmma_f32_16x16x32_bf16(false, aq[qa][0].v, false, bk[1][0].v, (short)0, s1, false, false);
            s1 = __builtin_amdgcn_wmma_f32_16x16x32_bf16(false, aq[qa][1].v, false, bk[1][1].v, (short)0, s1, false, false);

            // online softmax (row = vgpr r + 8*half; cols across 16 lanes)
            float p0[8], p1[8];
            #pragma unroll
            for (int r = 0; r < 8; ++r) {
                const float a0 = s0[r] * 32.0f;
                const float a1 = s1[r] * 32.0f;
                float mx = fmaxf(a0, a1);
                #pragma unroll
                for (int off = 1; off < 16; off <<= 1)
                    mx = fmaxf(mx, __shfl_xor(mx, off, 32));
                const float mnew = fmaxf(mrow[qa][r], mx);
                const float f  = __expf(mrow[qa][r] - mnew);
                const float e0 = __expf(a0 - mnew);
                const float e1 = __expf(a1 - mnew);
                float sum = e0 + e1;
                #pragma unroll
                for (int off = 1; off < 16; off <<= 1)
                    sum += __shfl_xor(sum, off, 32);
                lrow[qa][r] = lrow[qa][r] * f + sum;
                mrow[qa][r] = mnew;
                p0[r] = e0; p1[r] = e1;
                acc[qa][0][r] *= f; acc[qa][1][r] *= f;
                acc[qa][2][r] *= f; acc[qa][3][r] *= f;
            }

            // reshape P: C/D layout -> A layout via LDS (single-wave WG:
            // barriers are S_NOP; DS ops from one wave stay in order)
            __syncthreads();
            #pragma unroll
            for (int r = 0; r < 8; ++r) {
                const int row = r + 8 * half;
                const unsigned int pk = cvt2(p0[r], p1[r]);
                ldsP[row * PROW + lm]      = (unsigned short)pk;
                ldsP[row * PROW + 16 + lm] = (unsigned short)(pk >> 16);
            }
            __syncthreads();
            Frag ap;
            load_fragA(ap, &ldsP[lm * PROW], 0, half);   // 2x ds_load_b128

            // O += P @ V (contraction over 32 keys; V^T rows in LDS)
            #pragma unroll
            for (int t = 0; t < 4; ++t) {
                Frag bv;
                load_frag2(bv, &ldsV[buf][(t * 16 + lm) * VSTRIDE + half * 16]);
                acc[qa][t] = __builtin_amdgcn_wmma_f32_16x16x32_bf16(
                    false, ap.v, false, bv.v, (short)0, acc[qa][t], false, false);
            }
        }
    }

    // ---- epilogue: out[b][s][h*64 + d] = acc / l ----
    #pragma unroll
    for (int qa = 0; qa < 2; ++qa)
        #pragma unroll
        for (int t = 0; t < 4; ++t)
            #pragma unroll
            for (int r = 0; r < 8; ++r) {
                const int s = q0 + qa * 16 + r + 8 * half;
                out[((size_t)b * SEQ + s) * DMODEL + h * DH + t * 16 + lm] =
                    acc[qa][t][r] / lrow[qa][r];
            }
}

// ---------------------------------------------------------------------------
extern "C" void kernel_launch(void* const* d_in, const int* in_sizes, int n_in,
                              void* d_out, int out_size, void* d_ws, size_t ws_size,
                              hipStream_t stream) {
    (void)in_sizes; (void)n_in; (void)out_size; (void)ws_size;
    const float* X    = (const float*)d_in[0];   // query [B,S,D]
    const float* W    = (const float*)d_in[3];   // W_qkv_w [3D,D]
    const float* bias = (const float*)d_in[4];   // W_qkv_b [3D]
    float* O = (float*)d_out;

    const size_t headElems = (size_t)BATCH * NH * SEQ * DH;   // 4,194,304
    const size_t xElems    = (size_t)ROWS * DMODEL;           // 4,194,304
    const size_t wElems    = (size_t)NCOLS * DMODEL;          // 3,145,728
    unsigned short* wsq  = (unsigned short*)d_ws;
    unsigned short* wsk  = wsq  + headElems;
    unsigned short* wsvt = wsk  + headElems;
    unsigned short* xbf  = wsvt + headElems;
    unsigned short* wbf  = xbf  + xElems;        // total ~39.8 MB

    cvt_bf16_kernel<<<dim3((unsigned)(xElems / 4 / 256)), dim3(256), 0, stream>>>(
        X, xbf, (int)(xElems / 4));
    cvt_bf16_kernel<<<dim3((unsigned)(wElems / 4 / 256)), dim3(256), 0, stream>>>(
        W, wbf, (int)(wElems / 4));

    qkv_proj_kernel<<<dim3((ROWS / 32) * (NCOLS / 64) / 8), dim3(256), 0, stream>>>(
        xbf, wbf, bias, wsq, wsk, wsvt);

    // One wave per (b, h, 32-query block) -> 2048 blocks of 32 threads.
    attn_kernel<<<dim3(BATCH * NH * (SEQ / 32)), dim3(32), 0, stream>>>(
        wsq, wsk, wsvt, O);
}
